// CausalAttention_88098369176202
// MI455X (gfx1250) — compile-verified, matrix operand order checked by reading, not run on hardware
//
#include <hip/hip_runtime.h>
#include <cstdint>

// ---------- CDNA5 WMMA types ----------
typedef __attribute__((ext_vector_type(16))) __bf16 v16bf;
typedef __attribute__((ext_vector_type(8)))  float  v8f;

union BFrag {
    v16bf  v;
    __bf16 e[16];
    uint4  q[2];     // two 16-byte chunks per lane
};

union Pack8 {
    uint4  q;
    __bf16 e[8];
};

__device__ __forceinline__ v8f wmma_bf16(const BFrag& a, const BFrag& b, v8f c) {
    return __builtin_amdgcn_wmma_f32_16x16x32_bf16(
        false, a.v, false, b.v, (short)0, c, false, false);
}

// Per-lane fragment chunks (ISA 7.12.2, 16-bit A/B):
//   lanes 0-15 : k = {0..7, 16..23};  lanes 16-31 : k = {8..15, 24..31}
template <typename P>
__device__ __forceinline__ void load_frag(BFrag& f, const P* base, int half) {
    f.q[0] = *(const uint4*)(base + 8 * half);
    f.q[1] = *(const uint4*)(base + 16 + 8 * half);
}

// ---------- async 16B global->LDS copy (gfx1250 async path) ----------
typedef int v4i_ __attribute__((vector_size(16)));

__device__ __forceinline__ void async_copy16(const __bf16* g, __bf16* l) {
#if __has_builtin(__builtin_amdgcn_global_load_async_to_lds_b128)
    typedef __attribute__((address_space(1))) v4i_ gvec;
    typedef __attribute__((address_space(3))) v4i_ lvec;
    __builtin_amdgcn_global_load_async_to_lds_b128(
        (gvec*)(uintptr_t)g, (lvec*)(unsigned)(uintptr_t)l, 0, 0);
#else
    *(uint4*)l = *(const uint4*)g;     // fallback: direct copy
#endif
}
__device__ __forceinline__ void wait_async_le3() {
    asm volatile("s_wait_asynccnt 0x3" ::: "memory");
}
__device__ __forceinline__ void wait_async_0() {
    asm volatile("s_wait_asynccnt 0x0" ::: "memory");
}

// ---------- problem constants ----------
#define BATCH 2
#define SEQ   2048
#define CDIM  1024
#define NH    16
#define HD    64
#define MTOT  (BATCH * SEQ)     // 4096
#define LDT   40                // LDS row stride (80B: 16B-aligned, conflict-free)
#define NKT   (CDIM / 32)       // 32 K-steps

// ============================================================
// Prep kernels: one-time convert / convert+transpose into bf16
// ============================================================
__global__ __launch_bounds__(256)
void cvt_x_kernel(const float* __restrict__ X, __bf16* __restrict__ Xb) {
    size_t idx = ((size_t)blockIdx.x * 256 + threadIdx.x) * 8;
    float4 f0 = *(const float4*)&X[idx];
    float4 f1 = *(const float4*)&X[idx + 4];
    Pack8 p;
    p.e[0] = (__bf16)f0.x; p.e[1] = (__bf16)f0.y; p.e[2] = (__bf16)f0.z; p.e[3] = (__bf16)f0.w;
    p.e[4] = (__bf16)f1.x; p.e[5] = (__bf16)f1.y; p.e[6] = (__bf16)f1.z; p.e[7] = (__bf16)f1.w;
    *(uint4*)&Xb[idx] = p.q;
}

// W: [CDIM][N] fp32 row-major  ->  Wt: [N][CDIM] bf16 (n-major)
__global__ __launch_bounds__(256)
void cvt_transpose_w_kernel(const float* __restrict__ W,
                            __bf16* __restrict__ Wt, int N) {
    int n  = blockIdx.x * 256 + threadIdx.x;   // coalesced reads along n
    int kb = blockIdx.y;                       // 8 k's per thread
    Pack8 p;
    #pragma unroll
    for (int j = 0; j < 8; ++j)
        p.e[j] = (__bf16)W[(size_t)(kb * 8 + j) * N + n];
    *(uint4*)&Wt[(size_t)n * CDIM + kb * 8] = p.q;
}

// ============================================================
// Shared bf16 GEMM mainloop: A [.,CDIM] row-major, B [N,CDIM] n-major,
// 128x64 C-tile, double-buffered LDS fed by async global->LDS copies.
// ============================================================
__device__ __forceinline__ void gemm_mainloop(
    const __bf16* __restrict__ Ag, const __bf16* __restrict__ Bg,
    int m0, int n0,
    __bf16* lA0, __bf16* lA1, __bf16* lB0, __bf16* lB1,
    int tid, int ln, int half, int mg, int ng,
    v8f& acc00, v8f& acc01, v8f& acc10, v8f& acc11) {
    const int r  = tid >> 2;          // 0..63
    const int cc = (tid & 3) * 8;     // 16B chunk within 32-wide k-tile

    auto issue = [&](int k0, __bf16* lA, __bf16* lB) {
        async_copy16(&Ag[(size_t)(m0 + r) * CDIM + k0 + cc],      &lA[r * LDT + cc]);
        async_copy16(&Ag[(size_t)(m0 + r + 64) * CDIM + k0 + cc], &lA[(r + 64) * LDT + cc]);
        async_copy16(&Bg[(size_t)(n0 + r) * CDIM + k0 + cc],      &lB[r * LDT + cc]);
    };

    issue(0, lA0, lB0);
    for (int i = 0; i < NKT; ++i) {
        __bf16* cA = (i & 1) ? lA1 : lA0;
        __bf16* cB = (i & 1) ? lB1 : lB0;
        if (i + 1 < NKT) {
            issue((i + 1) * 32, (i & 1) ? lA0 : lA1, (i & 1) ? lB0 : lB1);
            wait_async_le3();          // tile i complete (3 newer ops pending)
        } else {
            wait_async_0();
        }
        __syncthreads();               // tile i visible to all waves

        BFrag a0, a1, b0, b1;
        load_frag(a0, &cA[(mg * 32 + ln) * LDT], half);
        load_frag(a1, &cA[(mg * 32 + 16 + ln) * LDT], half);
        load_frag(b0, &cB[(ng * 32 + ln) * LDT], half);
        load_frag(b1, &cB[(ng * 32 + 16 + ln) * LDT], half);
        acc00 = wmma_bf16(a0, b0, acc00);
        acc01 = wmma_bf16(a0, b1, acc01);
        acc10 = wmma_bf16(a1, b0, acc10);
        acc11 = wmma_bf16(a1, b1, acc11);
        __syncthreads();               // reads done before next prefetch lands
    }
}

// ============================================================
// Kernel 1: QKV GEMM  Xb[4096,1024](bf16) @ Wqt^T + b
//   epilogue: Q,K -> [B,H,T,D] ; V -> [B,H,D,T] (transposed), all bf16
// ============================================================
__global__ __launch_bounds__(256)
void qkv_gemm_kernel(const __bf16* __restrict__ Xb,
                     const __bf16* __restrict__ Wqt,
                     const float* __restrict__ bias,
                     __bf16* __restrict__ Qb,
                     __bf16* __restrict__ Kb,
                     __bf16* __restrict__ Vt) {
    __shared__ __align__(16) __bf16 lA0[128 * LDT], lA1[128 * LDT];
    __shared__ __align__(16) __bf16 lB0[64 * LDT],  lB1[64 * LDT];

    const int tid  = threadIdx.x;
    const int lane = tid & 31;
    const int wid  = tid >> 5;
    const int half = lane >> 4;
    const int ln   = lane & 15;
    const int m0 = blockIdx.x * 128;
    const int n0 = blockIdx.y * 64;
    const int mg = wid & 3;
    const int ng = wid >> 2;

    v8f acc00 = {}, acc01 = {}, acc10 = {}, acc11 = {};
    gemm_mainloop(Xb, Wqt, m0, n0, lA0, lA1, lB0, lB1,
                  tid, ln, half, mg, ng, acc00, acc01, acc10, acc11);

    #pragma unroll
    for (int im = 0; im < 2; ++im) {
        #pragma unroll
        for (int in = 0; in < 2; ++in) {
            v8f acc = (im == 0) ? (in == 0 ? acc00 : acc01)
                                : (in == 0 ? acc10 : acc11);
            int n = n0 + (ng * 2 + in) * 16 + ln;
            float bv = bias[n];
            int which = n >> 10;          // 0=q 1=k 2=v
            int rem = n & 1023;
            int h = rem >> 6, d = rem & 63;
            #pragma unroll
            for (int r = 0; r < 8; ++r) {
                int m  = m0 + (mg * 2 + im) * 16 + r + 8 * half;
                int b  = m >> 11;
                int tp = m & (SEQ - 1);
                __bf16 val = (__bf16)(acc[r] + bv);
                if (which == 0)
                    Qb[(((size_t)(b * NH + h) * SEQ) + tp) * HD + d] = val;
                else if (which == 1)
                    Kb[(((size_t)(b * NH + h) * SEQ) + tp) * HD + d] = val;
                else
                    Vt[(((size_t)(b * NH + h) * HD) + d) * SEQ + tp] = val;
            }
        }
    }
}

// ============================================================
// Kernel 2: causal flash attention, one wave per 16-row Q tile
// ============================================================
__global__ __launch_bounds__(256)
void attn_kernel(const __bf16* __restrict__ Qb,
                 const __bf16* __restrict__ Kb,
                 const __bf16* __restrict__ Vt,
                 __bf16* __restrict__ Ob) {
    __shared__ __align__(16) __bf16 pstage[8 * 16 * 32];

    const int tid  = threadIdx.x;
    const int lane = tid & 31;
    const int wid  = tid >> 5;
    const int half = lane >> 4;
    const int ln   = lane & 15;

    const int gw = blockIdx.x * 8 + wid;
    const int qi = gw & 127;
    const int h  = (gw >> 7) & (NH - 1);
    const int b  = gw >> 11;
    const int q0 = qi * 16;

    const __bf16* Qh = Qb + (size_t)(b * NH + h) * SEQ * HD;
    const __bf16* Kh = Kb + (size_t)(b * NH + h) * SEQ * HD;
    const __bf16* Vh = Vt + (size_t)(b * NH + h) * HD * SEQ;
    __bf16* pb = pstage + wid * 512;

    BFrag qf0, qf1;
    load_frag(qf0, &Qh[(size_t)(q0 + ln) * HD], half);
    load_frag(qf1, &Qh[(size_t)(q0 + ln) * HD + 32], half);

    float mrow[8], lrow[8];
    v8f o0 = {}, o1 = {}, o2 = {}, o3 = {};
    #pragma unroll
    for (int r = 0; r < 8; ++r) { mrow[r] = -1e30f; lrow[r] = 0.f; }

    const int ntiles = (q0 + 15) / 32 + 1;
    for (int jt = 0; jt < ntiles; ++jt) {
        const int jbase = jt * 32;

        v8f s[2];
        #pragma unroll
        for (int hh = 0; hh < 2; ++hh) {
            int kb = jbase + hh * 16;
            BFrag kf0, kf1;
            load_frag(kf0, &Kh[(size_t)(kb + ln) * HD], half);
            load_frag(kf1, &Kh[(size_t)(kb + ln) * HD + 32], half);
            v8f sv = {};
            sv = wmma_bf16(qf0, kf0, sv);
            sv = wmma_bf16(qf1, kf1, sv);
            #pragma unroll
            for (int r = 0; r < 8; ++r) {
                int row = q0 + r + 8 * half;
                int key = kb + ln;
                float x = sv[r] * 0.125f;
                sv[r] = (key <= row) ? x : -1e30f;
            }
            s[hh] = sv;
        }

        float mnew[8], alpha[8];
        #pragma unroll
        for (int r = 0; r < 8; ++r) {
            float mx = fmaxf(s[0][r], s[1][r]);
            #pragma unroll
            for (int d = 1; d < 16; d <<= 1)
                mx = fmaxf(mx, __shfl_xor(mx, d, 32));
            mnew[r]  = fmaxf(mrow[r], mx);
            alpha[r] = __expf(mrow[r] - mnew[r]);
            mrow[r]  = mnew[r];
        }

        #pragma unroll
        for (int hh = 0; hh < 2; ++hh) {
            #pragma unroll
            for (int r = 0; r < 8; ++r) {
                float p = __expf(s[hh][r] - mnew[r]);
                pb[(r + 8 * half) * 32 + hh * 16 + ln] = (__bf16)p;
                float sum = p;
                #pragma unroll
                for (int d = 1; d < 16; d <<= 1)
                    sum += __shfl_xor(sum, d, 32);
                lrow[r] = (hh == 0) ? (lrow[r] * alpha[r] + sum)
                                    : (lrow[r] + sum);
            }
        }

        #pragma unroll
        for (int r = 0; r < 8; ++r) {
            o0[r] *= alpha[r]; o1[r] *= alpha[r];
            o2[r] *= alpha[r]; o3[r] *= alpha[r];
        }

        BFrag pf;
        load_frag(pf, &pb[ln * 32], half);

        #pragma unroll
        for (int nc = 0; nc < 4; ++nc) {
            BFrag vf;
            load_frag(vf, &Vh[(size_t)(nc * 16 + ln) * SEQ + jbase], half);
            v8f& oacc = (nc == 0) ? o0 : (nc == 1) ? o1 : (nc == 2) ? o2 : o3;
            oacc = wmma_bf16(pf, vf, oacc);
        }
    }

    #pragma unroll
    for (int nc = 0; nc < 4; ++nc) {
        const v8f& oacc = (nc == 0) ? o0 : (nc == 1) ? o1 : (nc == 2) ? o2 : o3;
        #pragma unroll
        for (int r = 0; r < 8; ++r) {
            int t = q0 + r + 8 * half;
            Ob[((size_t)(b * SEQ + t)) * CDIM + h * HD + nc * 16 + ln] =
                (__bf16)(oacc[r] / lrow[r]);
        }
    }
}

// ============================================================
// Kernel 3: out = Ob(bf16)[4096,1024] @ Wpt^T + b_proj  (fp32 out)
// ============================================================
__global__ __launch_bounds__(256)
void proj_gemm_kernel(const __bf16* __restrict__ A,
                      const __bf16* __restrict__ Wpt,
                      const float* __restrict__ bias,
                      float* __restrict__ out) {
    __shared__ __align__(16) __bf16 lA0[128 * LDT], lA1[128 * LDT];
    __shared__ __align__(16) __bf16 lB0[64 * LDT],  lB1[64 * LDT];

    const int tid  = threadIdx.x;
    const int lane = tid & 31;
    const int wid  = tid >> 5;
    const int half = lane >> 4;
    const int ln   = lane & 15;
    const int m0 = blockIdx.x * 128;
    const int n0 = blockIdx.y * 64;
    const int mg = wid & 3;
    const int ng = wid >> 2;

    v8f acc00 = {}, acc01 = {}, acc10 = {}, acc11 = {};
    gemm_mainloop(A, Wpt, m0, n0, lA0, lA1, lB0, lB1,
                  tid, ln, half, mg, ng, acc00, acc01, acc10, acc11);

    #pragma unroll
    for (int im = 0; im < 2; ++im) {
        #pragma unroll
        for (int in = 0; in < 2; ++in) {
            v8f acc = (im == 0) ? (in == 0 ? acc00 : acc01)
                                : (in == 0 ? acc10 : acc11);
            int n = n0 + (ng * 2 + in) * 16 + ln;
            float bv = bias[n];
            #pragma unroll
            for (int r = 0; r < 8; ++r) {
                int m = m0 + (mg * 2 + im) * 16 + r + 8 * half;
                out[(size_t)m * CDIM + n] = acc[r] + bv;
            }
        }
    }
}

// ============================================================
extern "C" void kernel_launch(void* const* d_in, const int* in_sizes, int n_in,
                              void* d_out, int out_size, void* d_ws, size_t ws_size,
                              hipStream_t stream) {
    const float* x     = (const float*)d_in[0];
    const float* Wqkv  = (const float*)d_in[1];
    const float* bqkv  = (const float*)d_in[2];
    const float* Wproj = (const float*)d_in[3];
    const float* bproj = (const float*)d_in[4];
    float* out = (float*)d_out;

    const size_t per = (size_t)BATCH * NH * SEQ * HD;   // 4,194,304
    __bf16* Qb  = (__bf16*)d_ws;
    __bf16* Kb  = Qb + per;
    __bf16* Vt  = Kb + per;                 // [B,H,D,T]
    __bf16* Ob  = Vt + per;                 // [B,T,C]
    __bf16* Xb  = Ob + per;                 // [4096,1024]
    __bf16* Wqt = Xb + (size_t)MTOT * CDIM; // [3072,1024] n-major
    __bf16* Wpt = Wqt + (size_t)3 * CDIM * CDIM;  // [1024,1024] n-major

    cvt_x_kernel<<<dim3((MTOT * CDIM) / (256 * 8)), 256, 0, stream>>>(x, Xb);
    cvt_transpose_w_kernel<<<dim3((3 * CDIM) / 256, CDIM / 8), 256, 0, stream>>>(
        Wqkv, Wqt, 3 * CDIM);
    cvt_transpose_w_kernel<<<dim3(CDIM / 256, CDIM / 8), 256, 0, stream>>>(
        Wproj, Wpt, CDIM);

    qkv_gemm_kernel<<<dim3(MTOT / 128, (3 * CDIM) / 64), 256, 0, stream>>>(
        Xb, Wqt, bqkv, Qb, Kb, Vt);

    attn_kernel<<<dim3((BATCH * NH * (SEQ / 16)) / 8), 256, 0, stream>>>(
        Qb, Kb, Vt, Ob);

    proj_gemm_kernel<<<dim3(MTOT / 128, CDIM / 64), 256, 0, stream>>>(
        Ob, Wpt, bproj, out);
}
